// test_LSTM_90623809945959
// MI455X (gfx1250) — matrix operand order repaired and checked by reading, not compiled, not run on hardware
//
#include <hip/hip_runtime.h>
#include <hip/hip_bf16.h>

// ---------------------------------------------------------------------------
// LSTM (SEQ=512, BATCH=64, EMBED=HIDDEN=1024) for MI455X / gfx1250 (wave32)
//
// Persistent-kernel design: the recurrence is a 512-long serial chain with
// only ~1 GFLOP per step, so per-step dispatch overhead dominates once the
// GEMM is on the WMMA pipe. A single kernel with 32 co-resident blocks loops
// over all timesteps, synchronizing with a device-scope atomic barrier
// (monotonic counter -> no reset hazard; reset per launch by lstm_init).
//
// Per step, block bk owns hidden units [bk*32, bk*32+32):
//   - 8 waves x (one 16-wide gate-column tile vs 4 batch M-tiles) via
//     v_wmma_f32_16x16x32_f16, software-pipelined k-loop (K = 2048 = [x|h]).
//   - gate tiles -> LDS (64x128 f32), __syncthreads, pointwise recurrence.
//   - cell state c lives in registers for the whole sequence (8 cells/thread).
//   - h -> out[s], f16(h) and f16(x_{s+1}) -> A16 ping-pong buffer.
//
// A16 double buffer: step s reads A[s&1], writes A[(s+1)&1]; one barrier per
// step at the end of pointwise keeps all cross-block hazards disjoint.
// Weights (16 MB f16) are L2-resident (192 MB L2); HBM traffic is just
// embeddings in + outputs out (~256 MB, ~11 us at 23.3 TB/s).
// ---------------------------------------------------------------------------

#define EMBED   1024
#define HIDDEN  1024
#define BATCH   64
#define SEQ     512
#define NGATES  4096   // 4*HIDDEN, PyTorch order: i, f, g, o
#define KDIM    2048   // EMBED + HIDDEN
#define NBLOCKS 32

typedef _Float16 half8  __attribute__((ext_vector_type(8)));
typedef _Float16 half16 __attribute__((ext_vector_type(16)));
typedef float    float8 __attribute__((ext_vector_type(8)));

__device__ __forceinline__ float sigmoidf_(float x) {
    return 1.0f / (1.0f + __expf(-x));
}

// B tile (32x16 f16): lane holds column n = lane&15, K = {8*kh..8*kh+7} and {16+8*kh..+7}
__device__ __forceinline__ half16 load_b16(const _Float16* p, int kh) {
    half8 lo = *(const half8*)(p + 8 * kh);
    half8 hi = *(const half8*)(p + 16 + 8 * kh);
    return __builtin_shufflevector(lo, hi, 0,1,2,3,4,5,6,7,8,9,10,11,12,13,14,15);
}

// A tile (16x32 f16): lane holds row m = lane&15,
// K = {8*kh..8*kh+7} and {16+8*kh..+7} -> two 16B chunks.
__device__ __forceinline__ half16 load_a16(const _Float16* arow, int k0, int kh) {
    half8 lo = *(const half8*)(arow + k0 + 8 * kh);
    half8 hi = *(const half8*)(arow + k0 + 16 + 8 * kh);
    return __builtin_shufflevector(lo, hi, 0,1,2,3,4,5,6,7,8,9,10,11,12,13,14,15);
}

// ---------------------------------------------------------------------------
// One-time (per launch) weight conversion: W16[n][k] f16
// ---------------------------------------------------------------------------
__global__ __launch_bounds__(256) void lstm_wconv(
    const float* __restrict__ W_ih, const float* __restrict__ W_hh,
    _Float16* __restrict__ W16)
{
    int idx = blockIdx.x * 256 + threadIdx.x;          // [0, NGATES*KDIM)
    int n = idx >> 11;                                 // / KDIM
    int k = idx & (KDIM - 1);
    float v = (k < EMBED) ? W_ih[n * EMBED + k]
                          : W_hh[n * HIDDEN + (k - EMBED)];
    W16[idx] = (_Float16)v;
}

// ---------------------------------------------------------------------------
// Init: barrier counter = 0, A16[0] = [f16(x_0) | 0]
// ---------------------------------------------------------------------------
__global__ __launch_bounds__(256) void lstm_init(
    const float* __restrict__ embed,   // [SEQ][BATCH][EMBED]
    _Float16* __restrict__ A16,        // [BATCH][KDIM] (buffer 0)
    unsigned* __restrict__ barrier_ctr)
{
    int idx = blockIdx.x * 256 + threadIdx.x;          // [0, BATCH*HIDDEN)
    int b = idx >> 10;
    int j = idx & (HIDDEN - 1);
    A16[b * KDIM + EMBED + j] = (_Float16)0.0f;
    A16[b * KDIM + j]         = (_Float16)embed[idx];  // s = 0 slice
    if (idx == 0) *barrier_ctr = 0u;
}

// ---------------------------------------------------------------------------
// Persistent kernel: all 512 timesteps, device-scope barrier between steps
// ---------------------------------------------------------------------------
__global__ __launch_bounds__(256) void lstm_persistent(
    _Float16* __restrict__ A16a,        // [BATCH][KDIM] ping
    _Float16* __restrict__ A16b,        // [BATCH][KDIM] pong
    const _Float16* __restrict__ W16,   // [NGATES][KDIM]
    const float* __restrict__ b_ih, const float* __restrict__ b_hh,
    const float* __restrict__ embed,    // [SEQ][BATCH][EMBED]
    float* __restrict__ out,            // [SEQ][BATCH][HIDDEN] + h + c tail
    unsigned* __restrict__ barrier_ctr)
{
    __shared__ float lgates[BATCH * 128];   // [m][g*32 + jj], 32 KB

    const int bk   = blockIdx.x;            // hidden slice [bk*32, bk*32+32)
    const int lane = threadIdx.x & 31;
    const int w    = threadIdx.x >> 5;      // 0..7: gate g = w>>1, half = w&1
    const int nl   = lane & 15;
    const int kh   = lane >> 4;

    // GEMM-phase constants
    const int g       = w >> 1;
    const int colbase = g * HIDDEN + bk * 32 + (w & 1) * 16;
    const _Float16* __restrict__ wrow = W16 + (size_t)(colbase + nl) * KDIM;

    // pointwise-phase constants: thread owns cells (m = w + 8*i, j = jglob)
    const int jj    = threadIdx.x & 31;
    const int jglob = bk * 32 + jj;
    const float bi = b_ih[jglob             ] + b_hh[jglob             ];
    const float bf = b_ih[jglob + 1 * HIDDEN] + b_hh[jglob + 1 * HIDDEN];
    const float bg = b_ih[jglob + 2 * HIDDEN] + b_hh[jglob + 2 * HIDDEN];
    const float bo = b_ih[jglob + 3 * HIDDEN] + b_hh[jglob + 3 * HIDDEN];
    float creg[8];
#pragma unroll
    for (int i = 0; i < 8; ++i) creg[i] = 0.0f;

    for (int s = 0; s < SEQ; ++s) {
        const _Float16* __restrict__ Acur = (s & 1) ? A16b : A16a;
        _Float16* __restrict__       Anxt = (s & 1) ? A16a : A16b;

        // ---- GEMM phase: 8 gate tiles x 4 M-tiles, K = 2048 -------------
        const _Float16* __restrict__ a0 = Acur + (size_t)( 0 + nl) * KDIM;
        const _Float16* __restrict__ a1 = Acur + (size_t)(16 + nl) * KDIM;
        const _Float16* __restrict__ a2 = Acur + (size_t)(32 + nl) * KDIM;
        const _Float16* __restrict__ a3 = Acur + (size_t)(48 + nl) * KDIM;

        float8 acc0 = {}, acc1 = {}, acc2 = {}, acc3 = {};

        half16 bt  = load_b16(wrow, kh);
        half16 at0 = load_a16(a0, 0, kh);
        half16 at1 = load_a16(a1, 0, kh);
        half16 at2 = load_a16(a2, 0, kh);
        half16 at3 = load_a16(a3, 0, kh);

        for (int k0 = 32; k0 < KDIM; k0 += 32) {
            __builtin_prefetch(wrow + k0 + 224, 0, 3);
            half16 btn  = load_b16(wrow + k0, kh);
            half16 atn0 = load_a16(a0, k0, kh);
            half16 atn1 = load_a16(a1, k0, kh);
            half16 atn2 = load_a16(a2, k0, kh);
            half16 atn3 = load_a16(a3, k0, kh);

            acc0 = __builtin_amdgcn_wmma_f32_16x16x32_f16(false, at0, false, bt, (short)0, acc0, false, false);
            acc1 = __builtin_amdgcn_wmma_f32_16x16x32_f16(false, at1, false, bt, (short)0, acc1, false, false);
            acc2 = __builtin_amdgcn_wmma_f32_16x16x32_f16(false, at2, false, bt, (short)0, acc2, false, false);
            acc3 = __builtin_amdgcn_wmma_f32_16x16x32_f16(false, at3, false, bt, (short)0, acc3, false, false);

            bt = btn; at0 = atn0; at1 = atn1; at2 = atn2; at3 = atn3;
        }
        acc0 = __builtin_amdgcn_wmma_f32_16x16x32_f16(false, at0, false, bt, (short)0, acc0, false, false);
        acc1 = __builtin_amdgcn_wmma_f32_16x16x32_f16(false, at1, false, bt, (short)0, acc1, false, false);
        acc2 = __builtin_amdgcn_wmma_f32_16x16x32_f16(false, at2, false, bt, (short)0, acc2, false, false);
        acc3 = __builtin_amdgcn_wmma_f32_16x16x32_f16(false, at3, false, bt, (short)0, acc3, false, false);

        // C/D layout: VGPR r holds (M = r + 8*kh, N = nl)
        {
            const int lcol = w * 16 + nl;
#pragma unroll
            for (int r = 0; r < 8; ++r) {
                int mrow = r + 8 * kh;
                lgates[( 0 + mrow) * 128 + lcol] = acc0[r];
                lgates[(16 + mrow) * 128 + lcol] = acc1[r];
                lgates[(32 + mrow) * 128 + lcol] = acc2[r];
                lgates[(48 + mrow) * 128 + lcol] = acc3[r];
            }
        }

        __syncthreads();

        // ---- pointwise recurrence (c in registers) -----------------------
#pragma unroll
        for (int i = 0; i < 8; ++i) {
            const int m   = w + 8 * i;            // batch row
            const int idx = m * HIDDEN + jglob;

            float giv = lgates[m * 128 +  0 + jj] + bi;
            float gfv = lgates[m * 128 + 32 + jj] + bf;
            float ggv = lgates[m * 128 + 64 + jj] + bg;
            float gov = lgates[m * 128 + 96 + jj] + bo;

            float iv = sigmoidf_(giv);
            float fv = sigmoidf_(gfv);
            float gv = tanhf(ggv);
            float ov = sigmoidf_(gov);

            float cn = fv * creg[i] + iv * gv;
            float hn = ov * tanhf(cn);
            creg[i] = cn;

            out[(size_t)s * (BATCH * HIDDEN) + idx] = hn;
            Anxt[m * KDIM + EMBED + jglob] = (_Float16)hn;           // h16 for s+1
            if (s + 1 < SEQ)                                         // x16 for s+1
                Anxt[m * KDIM + jglob] =
                    (_Float16)embed[(size_t)(s + 1) * (BATCH * EMBED) + idx];

            if (s == SEQ - 1) {                                      // (h, c) tail
                const size_t base = (size_t)SEQ * BATCH * HIDDEN;
                out[base + idx]                  = hn;
                out[base + BATCH * HIDDEN + idx] = cn;
            }
        }

        // ---- device-scope barrier (also covers LDS reuse next step) -----
        __threadfence();                       // release Anxt writes
        __syncthreads();
        if (threadIdx.x == 0) {
            __hip_atomic_fetch_add(barrier_ctr, 1u, __ATOMIC_ACQ_REL,
                                   __HIP_MEMORY_SCOPE_AGENT);
            const unsigned target = (unsigned)NBLOCKS * (unsigned)(s + 1);
            while (__hip_atomic_load(barrier_ctr, __ATOMIC_ACQUIRE,
                                     __HIP_MEMORY_SCOPE_AGENT) < target)
                __builtin_amdgcn_s_sleep(1);
        }
        __syncthreads();
        __threadfence();                       // acquire for next step's reads
    }
}

extern "C" void kernel_launch(void* const* d_in, const int* in_sizes, int n_in,
                              void* d_out, int out_size, void* d_ws, size_t ws_size,
                              hipStream_t stream) {
    // setup_inputs() order: input(int64, unused), input_embed, W_ih, W_hh, b_ih, b_hh
    const float* embed = (const float*)d_in[1];
    const float* W_ih  = (const float*)d_in[2];
    const float* W_hh  = (const float*)d_in[3];
    const float* b_ih  = (const float*)d_in[4];
    const float* b_hh  = (const float*)d_in[5];
    float* out = (float*)d_out;

    // Workspace carve-up (~16.5 MB total)
    char* ws = (char*)d_ws;
    _Float16* W16  = (_Float16*)ws;  ws += (size_t)NGATES * KDIM * sizeof(_Float16); // 16 MB
    _Float16* A16a = (_Float16*)ws;  ws += (size_t)BATCH * KDIM * sizeof(_Float16);  // 256 KB
    _Float16* A16b = (_Float16*)ws;  ws += (size_t)BATCH * KDIM * sizeof(_Float16);  // 256 KB
    unsigned* bctr = (unsigned*)ws;  ws += 128;                                      // barrier

    // 1) f32 -> f16 weight conversion (re-done every launch: deterministic)
    lstm_wconv<<<(NGATES * KDIM) / 256, 256, 0, stream>>>(W_ih, W_hh, W16);

    // 2) stage x_0 into A16 buffer 0, reset barrier counter
    lstm_init<<<(BATCH * HIDDEN) / 256, 256, 0, stream>>>(embed, A16a, bctr);

    // 3) single persistent kernel runs all 512 steps
    lstm_persistent<<<NBLOCKS, 256, 0, stream>>>(A16a, A16b, W16, b_ih, b_hh,
                                                 embed, out, bctr);
}
